// MultiHeadSelfAttention_23261542875388
// MI455X (gfx1250) — matrix operand order repaired and checked by reading, not compiled, not run on hardware
//
#include <hip/hip_runtime.h>
#include <stdint.h>

// Problem constants (match reference)
#define SEQ   2048
#define DM    1024
#define NH    16
#define DK    64
#define BATCH 2
#define MROWS (BATCH*SEQ)   // 4096

typedef __attribute__((ext_vector_type(16))) __bf16 v16bf;
typedef __attribute__((ext_vector_type(8)))  float  v8f;
typedef __attribute__((ext_vector_type(4)))  unsigned int u32x4;
typedef __attribute__((ext_vector_type(8)))  unsigned int u32x8;

union Frag {
  v16bf bf;
  u32x4 q[2];
};

__device__ __forceinline__ uint16_t f2bf(float f) {
  unsigned x = __float_as_uint(f);
  unsigned r = x + 0x7fffu + ((x >> 16) & 1u);   // round-to-nearest-even
  return (uint16_t)(r >> 16);
}

__device__ __forceinline__ v8f zero8() {
  v8f z;
#pragma unroll
  for (int i = 0; i < 8; ++i) z[i] = 0.0f;
  return z;
}

__device__ __forceinline__ v8f wmma_bf16(const Frag& a, const Frag& b, v8f c) {
  // (neg_a, A, neg_b, B, c_mod, C, reuse_a, reuse_b)
  return __builtin_amdgcn_wmma_f32_16x16x32_bf16(false, a.bf, false, b.bf,
                                                 (short)0, c, false, false);
}

// LDS byte offset of a __shared__-derived pointer (flat LDS addr low 32 bits)
__device__ __forceinline__ uint32_t lds_off(const void* p) {
  return (uint32_t)(uintptr_t)p;
}

// ---------------------------------------------------------------------------
// Tensor Data Mover: async 2-D tile load (global -> LDS), bf16 elements.
// Builds D# group0 (4 SGPRs) + group1 (8 SGPRs) per CDNA5 ISA
// and issues TENSOR_LOAD_TO_LDS (2-descriptor-group form; VADDR2/3 = NULL).
// Tracked with TENSORcnt; wait via __builtin_amdgcn_s_wait_tensorcnt.
// tile_d0 elements along contiguous dim0, tile_d1 rows; d0_stride in elements.
// ---------------------------------------------------------------------------
__device__ __forceinline__ void tdm_load_tile2d(const void* gptr, uint32_t ldsByteOff,
                                                uint32_t tensor_d0, uint32_t tensor_d1,
                                                uint32_t tile_d0, uint32_t tile_d1,
                                                uint64_t d0_stride) {
  const uint64_t ga = (uint64_t)gptr;
  u32x4 g0;
  g0[0] = 1u;                                               // count=1 (valid), load
  g0[1] = ldsByteOff;                                       // lds_addr
  g0[2] = (uint32_t)ga;                                     // global_addr[31:0]
  g0[3] = (uint32_t)((ga >> 32) & 0x1FFFFFFu) | (2u << 30); // addr[56:32] | type=2

  u32x8 g1;
  g1[0] = (1u << 16);                                       // data_size=1 -> 2 bytes
  g1[1] = (tensor_d0 & 0xFFFFu) << 16;                      // tensor_dim0[15:0]
  g1[2] = (tensor_d0 >> 16) | ((tensor_d1 & 0xFFFFu) << 16);
  g1[3] = (tensor_d1 >> 16) | (tile_d0 << 16);              // tile_dim0
  g1[4] = (tile_d1 & 0xFFFFu);                              // tile_dim1 (tile_dim2=0)
  g1[5] = (uint32_t)(d0_stride & 0xFFFFFFFFu);              // tensor_dim0_stride lo
  g1[6] = (uint32_t)(d0_stride >> 32);                      // stride hi | dim1_stride lo16=0
  g1[7] = 0u;                                               // dim1_stride hi = 0

  asm volatile("tensor_load_to_lds %0, %1" :: "s"(g0), "s"(g1) : "memory");
}

// ---------------------------------------------------------------------------
// Elementwise casts
// ---------------------------------------------------------------------------
__global__ void k_cast_bf16(const float* __restrict__ in,
                            uint16_t* __restrict__ out, int n) {
  int i = blockIdx.x * blockDim.x + threadIdx.x;
  if (i < n) out[i] = f2bf(in[i]);
}

// in[K][N] (row-major) -> out[N][K] (row-major), fp32 -> bf16
__global__ void k_transpose_cast(const float* __restrict__ in,
                                 uint16_t* __restrict__ out, int K, int N) {
  int i = blockIdx.x * blockDim.x + threadIdx.x;
  if (i >= K * N) return;
  int k = i / N;
  int n = i - k * N;
  out[(size_t)n * K + k] = f2bf(in[i]);
}

// ---------------------------------------------------------------------------
// Block-level WMMA mainloop with TDM-staged, double-buffered B tile in LDS.
//   A:  [M x K] bf16 row-major (per-wave direct global b128 loads, ping-pong)
//   BT: [N x K] bf16 row-major (B transposed); block tile = 64 cols x 32 k
//   Wave computes rows [rowBase,rowBase+16) x cols [colBase,colBase+64).
// ---------------------------------------------------------------------------
__device__ __forceinline__ void gemm_block(const uint16_t* __restrict__ A, int lda,
                                           const uint16_t* __restrict__ BT, int ldb,
                                           int Ntotal, int rowBase, int colBase, int K,
                                           uint16_t (*ldsB)[64][32],  // [2][64][32]
                                           v8f* acc) {
  const int warp = threadIdx.x >> 5;
  const int lane = threadIdx.x & 31;
  const int half = lane >> 4;
  const int l16  = lane & 15;
  const uint16_t* arow = A + (size_t)(rowBase + l16) * lda;

  // Prologue: stage first B tile
  if (warp == 0) {
    tdm_load_tile2d(BT + (size_t)colBase * ldb, lds_off(&ldsB[0][0][0]),
                    (uint32_t)ldb, (uint32_t)Ntotal, 32u, 64u, (uint64_t)ldb);
    __builtin_amdgcn_s_wait_tensorcnt(0);
  }
  __syncthreads();

  Frag fa[2];
  fa[0].q[0] = *(const u32x4*)(arow + half * 8);
  fa[0].q[1] = *(const u32x4*)(arow + 16 + half * 8);

  int cur = 0, p = 0;
  for (int k = 0; k < K; k += 32) {
    // async-stage next B tile into the alternate buffer
    if (warp == 0 && k + 32 < K) {
      tdm_load_tile2d(BT + (size_t)colBase * ldb + (k + 32),
                      lds_off(&ldsB[cur ^ 1][0][0]),
                      (uint32_t)ldb, (uint32_t)Ntotal, 32u, 64u, (uint64_t)ldb);
    }
    // ping-pong next A fragment (no register copies at iteration end)
    if (k + 32 < K) {
      fa[p ^ 1].q[0] = *(const u32x4*)(arow + k + 32 + half * 8);
      fa[p ^ 1].q[1] = *(const u32x4*)(arow + k + 48 + half * 8);
    }

    // batch all B-fragment LDS loads, then issue the WMMA pack
    Frag fb[4];
#pragma unroll
    for (int t = 0; t < 4; ++t) {
      const uint16_t* bp = &ldsB[cur][t * 16 + l16][half * 8];
      fb[t].q[0] = *(const u32x4*)(bp);
      fb[t].q[1] = *(const u32x4*)(bp + 16);
    }
#pragma unroll
    for (int t = 0; t < 4; ++t)
      acc[t] = wmma_bf16(fa[p], fb[t], acc[t]);

    if (warp == 0) __builtin_amdgcn_s_wait_tensorcnt(0);
    __syncthreads();
    cur ^= 1;
    p ^= 1;
  }
}

// ---------------------------------------------------------------------------
// QKV projection: [4096 x 1024] @ [1024 x 3072] + b_qkv
// Scatters into q[B,H,S,dk], k[B,H,S,dk], vT[B,H,dk,S] (all bf16)
// ---------------------------------------------------------------------------
__global__ void k_gemm_qkv(const uint16_t* __restrict__ xb,
                           const uint16_t* __restrict__ wT,   // [3072][1024]
                           const float* __restrict__ bias,
                           uint16_t* __restrict__ qb,
                           uint16_t* __restrict__ kb,
                           uint16_t* __restrict__ vT) {
  __shared__ __align__(16) uint16_t ldsB[2][64][32];

  const int warp = threadIdx.x >> 5;
  const int lane = threadIdx.x & 31;
  const int half = lane >> 4;
  const int l16  = lane & 15;
  const int colBase = blockIdx.x * 64;
  const int rowBase = blockIdx.y * 128 + warp * 16;

  v8f acc[4] = {zero8(), zero8(), zero8(), zero8()};
  gemm_block(xb, DM, wT, DM, 3 * DM, rowBase, colBase, DM, ldsB, acc);

#pragma unroll
  for (int t = 0; t < 4; ++t) {
    const int n = colBase + t * 16 + l16;   // 0..3071
    const float bv = bias[n];
    const int chunk = n >> 10;              // 0=q 1=k 2=v
    const int d = n & (DM - 1);
    const int h = d >> 6;
    const int f = d & (DK - 1);
#pragma unroll
    for (int r = 0; r < 8; ++r) {
      const int m = rowBase + r + 8 * half;       // global row: b*SEQ + s
      const int b = m >> 11;
      const int s = m & (SEQ - 1);
      const uint16_t val = f2bf(acc[t][r] + bv);
      const size_t bh = (size_t)b * NH + h;
      if (chunk == 0)       qb[(bh * SEQ + s) * DK + f] = val;
      else if (chunk == 1)  kb[(bh * SEQ + s) * DK + f] = val;
      else                  vT[(bh * DK + f) * SEQ + s] = val;
    }
  }
}

// ---------------------------------------------------------------------------
// Flash attention: block = 4 waves = 64 query rows of one (b,h).
// K/V tiles (32 keys) staged by TDM into double-buffered LDS, shared by waves.
// grid = (SEQ/64, BATCH*NH), block = 128
// ---------------------------------------------------------------------------
__global__ void k_attention(const uint16_t* __restrict__ qb,
                            const uint16_t* __restrict__ kb,
                            const uint16_t* __restrict__ vT,
                            uint16_t* __restrict__ ao)   // [MROWS][DM] bf16
{
  __shared__ __align__(16) uint16_t ldsK[2][32][64];   // [buf][key][feat]
  __shared__ __align__(16) uint16_t ldsV[2][64][32];   // [buf][feat][key]
  __shared__ __align__(16) uint16_t ldsP[4][16][32];   // per-wave P staging

  const int warp = threadIdx.x >> 5;
  const int lane = threadIdx.x & 31;
  const int half = lane >> 4;
  const int l16  = lane & 15;

  const int bh = blockIdx.y;                 // 0..31
  const int b  = bh >> 4;
  const int h  = bh & (NH - 1);
  const int qRowBase = blockIdx.x * 64 + warp * 16;

  const uint16_t* qh  = qb + (size_t)bh * SEQ * DK;
  const uint16_t* kh  = kb + (size_t)bh * SEQ * DK;
  const uint16_t* vTh = vT + (size_t)bh * DK * SEQ;

  // Q fragments: two K-steps covering d_k = 64
  Frag fq[2];
  {
    const uint16_t* qp = qh + (size_t)(qRowBase + l16) * DK;
#pragma unroll
    for (int s = 0; s < 2; ++s) {
      fq[s].q[0] = *(const u32x4*)(qp + s * 32 + half * 8);
      fq[s].q[1] = *(const u32x4*)(qp + s * 32 + 16 + half * 8);
    }
  }

  // Prologue: stage first K/V tiles
  if (warp == 0) {
    tdm_load_tile2d(kh, lds_off(&ldsK[0][0][0]), DK, SEQ, DK, 32u, DK);
    tdm_load_tile2d(vTh, lds_off(&ldsV[0][0][0]), SEQ, DK, 32u, DK, SEQ);
    __builtin_amdgcn_s_wait_tensorcnt(0);
  }
  __syncthreads();

  v8f acc[4] = {zero8(), zero8(), zero8(), zero8()};
  float mrow[8], lrow[8];
#pragma unroll
  for (int r = 0; r < 8; ++r) { mrow[r] = -__builtin_inff(); lrow[r] = 0.0f; }

  const float L2E = 1.4426950408889634f;   // log2(e)

  int cur = 0;
  for (int keyBase = 0; keyBase < SEQ; keyBase += 32) {
    // async-stage next K/V tiles
    if (warp == 0 && keyBase + 32 < SEQ) {
      tdm_load_tile2d(kh + (size_t)(keyBase + 32) * DK,
                      lds_off(&ldsK[cur ^ 1][0][0]), DK, SEQ, DK, 32u, DK);
      tdm_load_tile2d(vTh + (keyBase + 32),
                      lds_off(&ldsV[cur ^ 1][0][0]), SEQ, DK, 32u, DK, SEQ);
    }

    // ---- batch all K-fragment LDS loads (fk[s*2+j]) ----
    Frag fk[4];
#pragma unroll
    for (int s = 0; s < 2; ++s) {
#pragma unroll
      for (int j = 0; j < 2; ++j) {
        const uint16_t* kp = &ldsK[cur][j * 16 + l16][s * 32 + half * 8];
        fk[s * 2 + j].q[0] = *(const u32x4*)(kp);
        fk[s * 2 + j].q[1] = *(const u32x4*)(kp + 16);
      }
    }
    // ---- V-fragment loads issued early: latency hides under softmax VALU ----
    Frag fv[4];
#pragma unroll
    for (int t = 0; t < 4; ++t) {
      const uint16_t* vp = &ldsV[cur][t * 16 + l16][half * 8];
      fv[t].q[0] = *(const u32x4*)(vp);
      fv[t].q[1] = *(const u32x4*)(vp + 16);
    }

    // ---- scores: two 16x16 tiles (keys j=0: +0..15, j=1: +16..31) ----
    v8f st0 = zero8(), st1 = zero8();
    st0 = wmma_bf16(fq[0], fk[0], st0);
    st1 = wmma_bf16(fq[0], fk[1], st1);
    st0 = wmma_bf16(fq[1], fk[2], st0);
    st1 = wmma_bf16(fq[1], fk[3], st1);

    // ---- online softmax (scale 1/sqrt(64) = 0.125) ----
    float p0[8], p1[8];
#pragma unroll
    for (int r = 0; r < 8; ++r) {
      st0[r] *= 0.125f;
      st1[r] *= 0.125f;
      float rm = fmaxf(st0[r], st1[r]);
#pragma unroll
      for (int msk = 1; msk <= 8; msk <<= 1)
        rm = fmaxf(rm, __shfl_xor(rm, msk, 32));
      const float mnew = fmaxf(mrow[r], rm);
      const float corr = exp2f((mrow[r] - mnew) * L2E);
      mrow[r] = mnew;
      p0[r] = exp2f((st0[r] - mnew) * L2E);
      p1[r] = exp2f((st1[r] - mnew) * L2E);
      float ps = p0[r] + p1[r];
#pragma unroll
      for (int msk = 1; msk <= 8; msk <<= 1)
        ps += __shfl_xor(ps, msk, 32);
      lrow[r] = lrow[r] * corr + ps;
#pragma unroll
      for (int t = 0; t < 4; ++t) acc[t][r] *= corr;
    }

    // ---- P (C-layout) -> LDS -> A-fragment layout (per-wave buffer) ----
#pragma unroll
    for (int r = 0; r < 8; ++r) {
      ldsP[warp][r + 8 * half][l16]      = f2bf(p0[r]);
      ldsP[warp][r + 8 * half][16 + l16] = f2bf(p1[r]);
    }
    asm volatile("s_wait_dscnt 0" ::: "memory");
    Frag fp;
    {
      const uint16_t* pp = &ldsP[warp][l16][0];
      fp.q[0] = *(const u32x4*)(pp + half * 8);
      fp.q[1] = *(const u32x4*)(pp + 16 + half * 8);
    }
    asm volatile("" ::: "memory");

    // ---- acc += P @ V_tile : 4 feature column tiles of 16 ----
#pragma unroll
    for (int t = 0; t < 4; ++t)
      acc[t] = wmma_bf16(fp, fv[t], acc[t]);

    if (warp == 0) __builtin_amdgcn_s_wait_tensorcnt(0);
    __syncthreads();
    cur ^= 1;
  }

  // ---- epilogue: normalize and store to attn_out [MROWS][DM] bf16 ----
#pragma unroll
  for (int t = 0; t < 4; ++t) {
#pragma unroll
    for (int r = 0; r < 8; ++r) {
      const int row = b * SEQ + qRowBase + r + 8 * half;
      const int col = h * DK + t * 16 + l16;
      ao[(size_t)row * DM + col] = f2bf(acc[t][r] / lrow[r]);
    }
  }
}

// ---------------------------------------------------------------------------
// Output projection: [4096 x 1024] @ [1024 x 1024] + b_out -> fp32 d_out
// ---------------------------------------------------------------------------
__global__ void k_gemm_out(const uint16_t* __restrict__ ab,
                           const uint16_t* __restrict__ wT,   // [1024][1024]
                           const float* __restrict__ bias,
                           float* __restrict__ out) {
  __shared__ __align__(16) uint16_t ldsB[2][64][32];

  const int warp = threadIdx.x >> 5;
  const int lane = threadIdx.x & 31;
  const int half = lane >> 4;
  const int l16  = lane & 15;
  const int colBase = blockIdx.x * 64;
  const int rowBase = blockIdx.y * 128 + warp * 16;

  v8f acc[4] = {zero8(), zero8(), zero8(), zero8()};
  gemm_block(ab, DM, wT, DM, DM, rowBase, colBase, DM, ldsB, acc);

#pragma unroll
  for (int t = 0; t < 4; ++t) {
    const int n = colBase + t * 16 + l16;
    const float bv = bias[n];
#pragma unroll
    for (int r = 0; r < 8; ++r) {
      const int m = rowBase + r + 8 * half;
      out[(size_t)m * DM + n] = acc[t][r] + bv;
    }
  }
}

// ---------------------------------------------------------------------------
// Host-side launcher
// ---------------------------------------------------------------------------
extern "C" void kernel_launch(void* const* d_in, const int* in_sizes, int n_in,
                              void* d_out, int out_size, void* d_ws, size_t ws_size,
                              hipStream_t stream) {
  const float* x    = (const float*)d_in[0];   // [2,2048,1024]
  const float* Wqkv = (const float*)d_in[1];   // [1024,3072]
  const float* bqkv = (const float*)d_in[2];   // [3072]
  const float* Wout = (const float*)d_in[3];   // [1024,1024]
  const float* bout = (const float*)d_in[4];   // [1024]
  float* out = (float*)d_out;                  // [2,2048,1024] fp32

  // Workspace carve-up (needs 48 MiB)
  char* ws = (char*)d_ws;
  uint16_t* xb  = (uint16_t*)(ws + ((size_t)0  << 20));  // 8 MiB  x bf16 [4096][1024]
  uint16_t* wqT = (uint16_t*)(ws + ((size_t)8  << 20));  // 6 MiB  W_qkv^T bf16 [3072][1024]
  uint16_t* woT = (uint16_t*)(ws + ((size_t)14 << 20));  // 2 MiB  W_out^T bf16 [1024][1024]
  uint16_t* qb  = (uint16_t*)(ws + ((size_t)16 << 20));  // 8 MiB  q [B,H,S,dk]
  uint16_t* kb  = (uint16_t*)(ws + ((size_t)24 << 20));  // 8 MiB  k [B,H,S,dk]
  uint16_t* vT  = (uint16_t*)(ws + ((size_t)32 << 20));  // 8 MiB  v [B,H,dk,S]
  uint16_t* ao  = (uint16_t*)(ws + ((size_t)40 << 20));  // 8 MiB  attn out [4096][1024]

  // 1) casts / transposes
  k_cast_bf16<<<(MROWS * DM) / 256, 256, 0, stream>>>(x, xb, MROWS * DM);
  k_transpose_cast<<<(DM * 3 * DM) / 256, 256, 0, stream>>>(Wqkv, wqT, DM, 3 * DM);
  k_transpose_cast<<<(DM * DM) / 256, 256, 0, stream>>>(Wout, woT, DM, DM);

  // 2) QKV projection
  k_gemm_qkv<<<dim3((3 * DM) / 64, MROWS / 128), 256, 0, stream>>>(xb, wqT, bqkv, qb, kb, vT);

  // 3) flash attention
  k_attention<<<dim3(SEQ / 64, BATCH * NH), 128, 0, stream>>>(qb, kb, vT, ao);

  // 4) output projection
  k_gemm_out<<<dim3(DM / 64, MROWS / 128), 256, 0, stream>>>(ao, woT, bout, out);
}
